// HGCNLayer_75187697484268
// MI455X (gfx1250) — matrix operand order repaired and checked by reading, not compiled, not run on hardware
//
#include <hip/hip_runtime.h>

// ---------------- problem constants (fixed by reference) ----------------
constexpr int N_NODES = 50000;
constexpr int N_EDGES = 500000;
constexpr int D = 64;
#define MAXN 0.996f            // 1 - 4e-3 (proj boundary)
#define EPSMIN 1e-15f

typedef __attribute__((ext_vector_type(16))) _Float16 v16h;
typedef __attribute__((ext_vector_type(8)))  float    v8f;

// ---------------- small device helpers ----------------
__device__ inline float artanh_c(float z) {
    z = fminf(fmaxf(z, -0.99999f), 0.99999f);   // clip(+-1e-5) per reference
    return atanhf(z);
}
__device__ inline float gelu_tanh(float x) {
    const float k0 = 0.7978845608028654f;       // sqrt(2/pi)
    const float k1 = 0.044715f;
    float x3 = x * x * x;
    return 0.5f * x * (1.f + tanhf(k0 * (x + k1 * x3)));
}
__device__ inline void atomicMaxF(float* addr, float val) {
    if (val >= 0.f) atomicMax((int*)addr, __float_as_int(val));
    else            atomicMin((unsigned int*)addr, __float_as_uint(val));
}

// A fragment (16x32 f16, M=rows across lanes, ISA 7.12.2 16-bit A layout):
// lane<16 : M=lane,    K = kt*32 + [0..7] and [16..23]
// lane>=16: M=lane-16, K = kt*32 + [8..15] and [24..31]
__device__ inline v16h load_a_frag(const _Float16* act, int stride, int kt, int lane) {
    int m   = lane & 15;
    int klo = kt * 32 + ((lane < 16) ? 0 : 8);
    const _Float16* pl = act + m * stride + klo;
    const _Float16* ph = pl + 16;
    v16h v;
#pragma unroll
    for (int j = 0; j < 8; ++j) { v[j] = pl[j]; v[8 + j] = ph[j]; }
    return v;
}

// B fragment (32x16 f16, K x N). Computing Y = X * W^T with W [outF x inF]
// row-major => B[k][n] = W[n][k].
// lane<16 : N=lane,    K = kt*32 + [0..7] and [16..23]
// lane>=16: N=lane-16, K = kt*32 + [8..15] and [24..31]
__device__ inline v16h load_b_frag(const _Float16* Wh, int inF, int nt, int kt, int lane) {
    int n  = nt * 16 + (lane & 15);
    int k0 = kt * 32 + ((lane < 16) ? 0 : 8);
    const _Float16* p = Wh + n * inF + k0;
    v16h v;
#pragma unroll
    for (int j = 0; j < 16; ++j) v[j] = p[j + ((j >= 8) ? 8 : 0)];
    return v;
}

__device__ inline v8f wmma16(v16h a, v16h b, v8f c) {
    return __builtin_amdgcn_wmma_f32_16x16x32_f16(false, a, false, b, (short)0, c, false, false);
}

// ---------------- misc kernels ----------------
__global__ void f32_to_f16_kernel(const float* src, _Float16* dst, int n) {
    int i = blockIdx.x * blockDim.x + threadIdx.x;
    if (i < n) dst[i] = (_Float16)src[i];
}

__global__ void init_kernel(float* cnt, float* sum, float* vmax, float* var,
                            float* den1, float* num1, float* den10, float* num10) {
    int i = blockIdx.x * blockDim.x + threadIdx.x;
    if (i < N_NODES * D) {
        sum[i] = 0.f; var[i] = 0.f;
        den1[i] = 0.f; num1[i] = 0.f; den10[i] = 0.f; num10[i] = 0.f;
        vmax[i] = -3.402823466e38f;
        if (i < N_NODES) cnt[i] = 0.f;
    }
}

// hyp_bias = proj(expmap0(b))   (64 threads, one block)
__global__ void bias_kernel(const float* b, float* hb) {
    __shared__ float red[64];
    int t = threadIdx.x;
    float v = b[t];
    red[t] = v * v; __syncthreads();
#pragma unroll
    for (int o = 32; o > 0; o >>= 1) { if (t < o) red[t] += red[t + o]; __syncthreads(); }
    float n = fmaxf(sqrtf(red[0]), EPSMIN); __syncthreads();
    float e = tanhf(n) / n * v;
    red[t] = e * e; __syncthreads();
#pragma unroll
    for (int o = 32; o > 0; o >>= 1) { if (t < o) red[t] += red[t + o]; __syncthreads(); }
    float ne = fmaxf(sqrtf(red[0]), EPSMIN);
    hb[t] = (ne > MAXN) ? e * (MAXN / ne) : e;
}

__device__ inline float reduce64(float v, float* red) {
    int t = threadIdx.x;
    red[t] = v; __syncthreads();
#pragma unroll
    for (int o = 32; o > 0; o >>= 1) { if (t < o) red[t] += red[t + o]; __syncthreads(); }
    float r = red[0]; __syncthreads();
    return r;
}

// HypLinear + logmap0 : one 64-thread block per node
__global__ void hyplinear_kernel(const float* __restrict__ x, const float* __restrict__ W,
                                 const float* __restrict__ hb, float* __restrict__ xt) {
    __shared__ float xs[64];
    __shared__ float red[64];
    int node = blockIdx.x, t = threadIdx.x;
    float xv = x[node * 64 + t];
    xs[t] = xv; __syncthreads();
    float mx = 0.f;
    const float* wr = W + t * 64;
#pragma unroll 8
    for (int k = 0; k < 64; ++k) mx += xs[k] * wr[k];
    float xn2 = reduce64(xv * xv, red);
    float mn2 = reduce64(mx * mx, red);
    float xn = fmaxf(sqrtf(xn2), EPSMIN), mn = fmaxf(sqrtf(mn2), EPSMIN);
    float res = (mn2 == 0.f) ? 0.f : tanhf(mn / xn * artanh_c(xn)) * mx / mn;
    float rn = fmaxf(sqrtf(reduce64(res * res, red)), EPSMIN);
    if (rn > MAXN) res *= MAXN / rn;                     // proj
    float hbv = hb[t];
    float x2 = reduce64(res * res, red);
    float y2 = reduce64(hbv * hbv, red);
    float xy = reduce64(res * hbv, red);
    float num = (1.f + 2.f * xy + y2) * res + (1.f - x2) * hbv;
    float den = fmaxf(1.f + 2.f * xy + x2 * y2, EPSMIN);
    float h = num / den;
    float hn = fmaxf(sqrtf(reduce64(h * h, red)), EPSMIN);
    if (hn > MAXN) h *= MAXN / hn;                       // proj
    hn = fmaxf(sqrtf(reduce64(h * h, red)), EPSMIN);
    xt[node * 64 + t] = artanh_c(hn) / hn * h;           // logmap0
}

// ------------- edge MLP: 16 edges/wave, WMMA, writes xe + cnt/sum/max -------------
__global__ void __launch_bounds__(32)
edge_mlp_kernel(const float* __restrict__ xt, const int* __restrict__ adj,
                const _Float16* __restrict__ w1h, const float* __restrict__ b1,
                const _Float16* __restrict__ w2h, const float* __restrict__ b2,
                const _Float16* __restrict__ w3h, const float* __restrict__ b3,
                float* __restrict__ xe, float* __restrict__ cnt,
                float* __restrict__ sum, float* __restrict__ vmax) {
    __shared__ _Float16 feat[16 * 192];
    __shared__ _Float16 act1[16 * 128];
    __shared__ _Float16 act2[16 * 128];
    __shared__ int rIdx[16], sIdx[16];
    int lane = threadIdx.x;
    int e0 = blockIdx.x * 16;
    if (lane < 16) { sIdx[lane] = adj[e0 + lane]; rIdx[lane] = adj[N_EDGES + e0 + lane]; }
    __syncthreads();
    { // gather: 2 lanes per edge, 32 dims each
        int e = lane >> 1, dbase = (lane & 1) * 32;
        int rr = rIdx[e], ss = sIdx[e];
#pragma unroll 4
        for (int d = 0; d < 32; ++d) {
            float a = xt[rr * 64 + dbase + d];
            float bb = xt[ss * 64 + dbase + d];
            feat[e * 192 + dbase + d]        = (_Float16)a;
            feat[e * 192 + 64 + dbase + d]   = (_Float16)bb;
            feat[e * 192 + 128 + dbase + d]  = (_Float16)(bb - a);
        }
    }
    __syncthreads();
    int m0 = (lane < 16) ? 0 : 8, ncol = lane & 15;
    // layer 1: [16x192] @ [192x128]^T
    v16h a1[6];
#pragma unroll
    for (int kt = 0; kt < 6; ++kt) a1[kt] = load_a_frag(feat, 192, kt, lane);
#pragma unroll
    for (int nt = 0; nt < 8; ++nt) {
        v8f c = {0.f, 0.f, 0.f, 0.f, 0.f, 0.f, 0.f, 0.f};
#pragma unroll
        for (int kt = 0; kt < 6; ++kt) c = wmma16(a1[kt], load_b_frag(w1h, 192, nt, kt, lane), c);
        int n = nt * 16 + ncol; float bv = b1[n];
#pragma unroll
        for (int g = 0; g < 8; ++g) act1[(m0 + g) * 128 + n] = (_Float16)gelu_tanh(c[g] + bv);
    }
    __syncthreads();
    // layer 2: [16x128] @ [128x128]^T
    v16h a2[4];
#pragma unroll
    for (int kt = 0; kt < 4; ++kt) a2[kt] = load_a_frag(act1, 128, kt, lane);
#pragma unroll
    for (int nt = 0; nt < 8; ++nt) {
        v8f c = {0.f, 0.f, 0.f, 0.f, 0.f, 0.f, 0.f, 0.f};
#pragma unroll
        for (int kt = 0; kt < 4; ++kt) c = wmma16(a2[kt], load_b_frag(w2h, 128, nt, kt, lane), c);
        int n = nt * 16 + ncol; float bv = b2[n];
#pragma unroll
        for (int g = 0; g < 8; ++g) act2[(m0 + g) * 128 + n] = (_Float16)gelu_tanh(c[g] + bv);
    }
    __syncthreads();
    // layer 3: [16x128] @ [128x64]^T  -> xe + segment atomics
    v16h a3[4];
#pragma unroll
    for (int kt = 0; kt < 4; ++kt) a3[kt] = load_a_frag(act2, 128, kt, lane);
#pragma unroll
    for (int nt = 0; nt < 4; ++nt) {
        v8f c = {0.f, 0.f, 0.f, 0.f, 0.f, 0.f, 0.f, 0.f};
#pragma unroll
        for (int kt = 0; kt < 4; ++kt) c = wmma16(a3[kt], load_b_frag(w3h, 128, nt, kt, lane), c);
        int n = nt * 16 + ncol; float bv = b3[n];
#pragma unroll
        for (int g = 0; g < 8; ++g) {
            int m = m0 + g;
            float v = c[g] + bv;
            int rr = rIdx[m];
            xe[(size_t)(e0 + m) * 64 + n] = v;
            atomicAdd(&sum[rr * 64 + n], v);
            atomicMaxF(&vmax[rr * 64 + n], v);
        }
    }
    if (lane < 16) atomicAdd(&cnt[rIdx[lane]], 1.0f);
}

__global__ void mean_kernel(const float* sum, const float* cnt, float* mean) {
    int i = blockIdx.x * blockDim.x + threadIdx.x;
    if (i >= N_NODES * D) return;
    mean[i] = sum[i] / fmaxf(cnt[i >> 6], 1.f);
}

// per (edge, dim): variance + softmax numerators/denominators (t=1, t=10)
__global__ void edge_stats_kernel(const int* __restrict__ adj, const float* __restrict__ xe,
                                  const float* __restrict__ mean, const float* __restrict__ vmax,
                                  float* var, float* den1, float* num1, float* den10, float* num10) {
    int i = blockIdx.x * blockDim.x + threadIdx.x;
    if (i >= N_EDGES * D) return;
    int e = i >> 6, d = i & 63;
    int r = adj[N_EDGES + e];
    int gi = r * 64 + d;
    float v = xe[i];
    float dm = v - mean[gi];
    atomicAdd(&var[gi], dm * dm);
    float mx = vmax[gi];
    float e1  = expf(v - mx);
    float e10 = expf(10.f * (v - mx));     // segmax(10v) = 10*segmax(v)
    atomicAdd(&den1[gi], e1);   atomicAdd(&num1[gi], v * e1);
    atomicAdd(&den10[gi], e10); atomicAdd(&num10[gi], v * e10);
}

// ------------- agg MLP (16 nodes/wave, WMMA) + hyperbolic tail -------------
__global__ void __launch_bounds__(32)
agg_mlp_kernel(const float* __restrict__ mean, const float* __restrict__ var,
               const float* __restrict__ num1, const float* __restrict__ den1,
               const float* __restrict__ num10, const float* __restrict__ den10,
               const float* __restrict__ cnt,
               const _Float16* __restrict__ w1h, const float* __restrict__ b1,
               const _Float16* __restrict__ w2h, const float* __restrict__ b2,
               const _Float16* __restrict__ w3h, const float* __restrict__ b3,
               float* __restrict__ out) {
    __shared__ _Float16 fin[16 * 256];
    __shared__ _Float16 act1[16 * 128];
    __shared__ _Float16 act2[16 * 128];
    __shared__ float xagg[16 * 64];
    int lane = threadIdx.x;
    int n0 = blockIdx.x * 16;
    { // build [16 x 256] input tile: [mean | var/cnt | sm1 | sm10]
        int i = lane >> 1, dbase = (lane & 1) * 32;
        int node = n0 + i;
        float cv = fmaxf(cnt[node], 1.f);
#pragma unroll 4
        for (int d = 0; d < 32; ++d) {
            int dd = dbase + d, gi = node * 64 + dd;
            fin[i * 256 + dd]        = (_Float16)mean[gi];
            fin[i * 256 + 64 + dd]   = (_Float16)(var[gi] / cv);
            fin[i * 256 + 128 + dd]  = (_Float16)(num1[gi] / fmaxf(den1[gi], EPSMIN));
            fin[i * 256 + 192 + dd]  = (_Float16)(num10[gi] / fmaxf(den10[gi], EPSMIN));
        }
    }
    __syncthreads();
    int m0 = (lane < 16) ? 0 : 8, ncol = lane & 15;
    // layer 1: [16x256] @ [256x128]^T
    v16h a1[8];
#pragma unroll
    for (int kt = 0; kt < 8; ++kt) a1[kt] = load_a_frag(fin, 256, kt, lane);
#pragma unroll
    for (int nt = 0; nt < 8; ++nt) {
        v8f c = {0.f, 0.f, 0.f, 0.f, 0.f, 0.f, 0.f, 0.f};
#pragma unroll
        for (int kt = 0; kt < 8; ++kt) c = wmma16(a1[kt], load_b_frag(w1h, 256, nt, kt, lane), c);
        int n = nt * 16 + ncol; float bv = b1[n];
#pragma unroll
        for (int g = 0; g < 8; ++g) act1[(m0 + g) * 128 + n] = (_Float16)gelu_tanh(c[g] + bv);
    }
    __syncthreads();
    // layer 2
    v16h a2[4];
#pragma unroll
    for (int kt = 0; kt < 4; ++kt) a2[kt] = load_a_frag(act1, 128, kt, lane);
#pragma unroll
    for (int nt = 0; nt < 8; ++nt) {
        v8f c = {0.f, 0.f, 0.f, 0.f, 0.f, 0.f, 0.f, 0.f};
#pragma unroll
        for (int kt = 0; kt < 4; ++kt) c = wmma16(a2[kt], load_b_frag(w2h, 128, nt, kt, lane), c);
        int n = nt * 16 + ncol; float bv = b2[n];
#pragma unroll
        for (int g = 0; g < 8; ++g) act2[(m0 + g) * 128 + n] = (_Float16)gelu_tanh(c[g] + bv);
    }
    __syncthreads();
    // layer 3 -> xagg (f32)
    v16h a3[4];
#pragma unroll
    for (int kt = 0; kt < 4; ++kt) a3[kt] = load_a_frag(act2, 128, kt, lane);
#pragma unroll
    for (int nt = 0; nt < 4; ++nt) {
        v8f c = {0.f, 0.f, 0.f, 0.f, 0.f, 0.f, 0.f, 0.f};
#pragma unroll
        for (int kt = 0; kt < 4; ++kt) c = wmma16(a3[kt], load_b_frag(w3h, 128, nt, kt, lane), c);
        int n = nt * 16 + ncol; float bv = b3[n];
#pragma unroll
        for (int g = 0; g < 8; ++g) xagg[(m0 + g) * 64 + n] = c[g] + bv;
    }
    __syncthreads();
    // hyperbolic tail: out = proj(expmap0(relu(logmap0(proj(expmap0(xagg))))))
    if (lane < 16) {
        const float* row = xagg + lane * 64;
        int node = n0 + lane;
        float n2 = 0.f;
#pragma unroll 8
        for (int d = 0; d < 64; ++d) { float v = row[d]; n2 += v * v; }
        float nn = fmaxf(sqrtf(n2), EPSMIN);
        float t  = tanhf(nn) / nn;                          // expmap0 scale
        float hn = fmaxf(tanhf(nn), EPSMIN);                // ||expmap0(xagg)||
        float f1 = (hn > MAXN) ? MAXN / hn : 1.f;           // proj
        float q  = fmaxf(hn * f1, EPSMIN);                  // ||h_agg||
        float lf = artanh_c(q) / q;                         // logmap0 scale
        float s2 = 0.f;
#pragma unroll 8
        for (int d = 0; d < 64; ++d) {
            float u = fmaxf(lf * t * f1 * row[d], 0.f);     // relu
            s2 += u * u;
        }
        float un = fmaxf(sqrtf(s2), EPSMIN);
        float ef = tanhf(un) / un;                          // expmap0
        float on = fmaxf(tanhf(un), EPSMIN);
        float f2 = (on > MAXN) ? MAXN / on : 1.f;           // proj
#pragma unroll 8
        for (int d = 0; d < 64; ++d) {
            float u = fmaxf(lf * t * f1 * row[d], 0.f);
            out[node * 64 + d] = ef * f2 * u;
        }
    }
}

// ---------------- host launcher ----------------
extern "C" void kernel_launch(void* const* d_in, const int* in_sizes, int n_in,
                              void* d_out, int out_size, void* d_ws, size_t ws_size,
                              hipStream_t stream) {
    const float* x   = (const float*)d_in[0];
    const int*   adj = (const int*)d_in[1];
    const float* W   = (const float*)d_in[2];
    const float* b   = (const float*)d_in[3];
    const float* ew1 = (const float*)d_in[4];  const float* eb1 = (const float*)d_in[5];
    const float* ew2 = (const float*)d_in[6];  const float* eb2 = (const float*)d_in[7];
    const float* ew3 = (const float*)d_in[8];  const float* eb3 = (const float*)d_in[9];
    const float* aw1 = (const float*)d_in[10]; const float* ab1 = (const float*)d_in[11];
    const float* aw2 = (const float*)d_in[12]; const float* ab2 = (const float*)d_in[13];
    const float* aw3 = (const float*)d_in[14]; const float* ab3 = (const float*)d_in[15];
    float* out = (float*)d_out;

    char* ws = (char*)d_ws;
    size_t off = 0;
    auto take = [&](size_t bytes) -> void* {
        void* p = ws + off;
        off = (off + bytes + 255) & ~(size_t)255;
        return p;
    };
    const size_t ND = (size_t)N_NODES * D;
    const size_t ED = (size_t)N_EDGES * D;
    float*     hb    = (float*)take(64 * sizeof(float));
    float*     xt    = (float*)take(ND * sizeof(float));
    _Float16*  w1h   = (_Float16*)take(128 * 192 * sizeof(_Float16));
    _Float16*  w2h   = (_Float16*)take(128 * 128 * sizeof(_Float16));
    _Float16*  w3h   = (_Float16*)take(64 * 128 * sizeof(_Float16));
    _Float16*  g1h   = (_Float16*)take(128 * 256 * sizeof(_Float16));
    _Float16*  g2h   = (_Float16*)take(128 * 128 * sizeof(_Float16));
    _Float16*  g3h   = (_Float16*)take(64 * 128 * sizeof(_Float16));
    float*     xe    = (float*)take(ED * sizeof(float));
    float*     cnt   = (float*)take((size_t)N_NODES * sizeof(float));
    float*     sum   = (float*)take(ND * sizeof(float));
    float*     vmax  = (float*)take(ND * sizeof(float));
    float*     meanp = (float*)take(ND * sizeof(float));
    float*     varp  = (float*)take(ND * sizeof(float));
    float*     den1  = (float*)take(ND * sizeof(float));
    float*     num1  = (float*)take(ND * sizeof(float));
    float*     den10 = (float*)take(ND * sizeof(float));
    float*     num10 = (float*)take(ND * sizeof(float));
    (void)ws_size; (void)in_sizes; (void)n_in; (void)out_size;

    // weights -> f16 (L2-resident reuse by WMMA kernels)
    f32_to_f16_kernel<<<(128 * 192 + 255) / 256, 256, 0, stream>>>(ew1, w1h, 128 * 192);
    f32_to_f16_kernel<<<(128 * 128 + 255) / 256, 256, 0, stream>>>(ew2, w2h, 128 * 128);
    f32_to_f16_kernel<<<(64  * 128 + 255) / 256, 256, 0, stream>>>(ew3, w3h, 64 * 128);
    f32_to_f16_kernel<<<(128 * 256 + 255) / 256, 256, 0, stream>>>(aw1, g1h, 128 * 256);
    f32_to_f16_kernel<<<(128 * 128 + 255) / 256, 256, 0, stream>>>(aw2, g2h, 128 * 128);
    f32_to_f16_kernel<<<(64  * 128 + 255) / 256, 256, 0, stream>>>(aw3, g3h, 64 * 128);

    bias_kernel<<<1, 64, 0, stream>>>(b, hb);
    hyplinear_kernel<<<N_NODES, 64, 0, stream>>>(x, W, hb, xt);

    init_kernel<<<(int)((ND + 255) / 256), 256, 0, stream>>>(cnt, sum, vmax, varp,
                                                             den1, num1, den10, num10);

    edge_mlp_kernel<<<N_EDGES / 16, 32, 0, stream>>>(xt, adj, w1h, eb1, w2h, eb2, w3h, eb3,
                                                     xe, cnt, sum, vmax);

    mean_kernel<<<(int)((ND + 255) / 256), 256, 0, stream>>>(sum, cnt, meanp);

    edge_stats_kernel<<<(int)((ED + 255) / 256), 256, 0, stream>>>(adj, xe, meanp, vmax,
                                                                   varp, den1, num1, den10, num10);

    agg_mlp_kernel<<<N_NODES / 16, 32, 0, stream>>>(meanp, varp, num1, den1, num10, den10, cnt,
                                                    g1h, ab1, g2h, ab2, g3h, ab3, out);
}
